// CrossAttentionContextGenerator_38087769981257
// MI455X (gfx1250) — matrix operand order repaired and checked by reading, not compiled, not run on hardware
//
#include <hip/hip_runtime.h>
#include <hip/hip_bf16.h>

// ---------------------------------------------------------------------------
// Types for CDNA5 WMMA
// ---------------------------------------------------------------------------
typedef __attribute__((ext_vector_type(16))) _Float16 v16h;
typedef __attribute__((ext_vector_type(8)))  float    v8f;
typedef __attribute__((ext_vector_type(4)))  float    fvec4;
typedef __attribute__((ext_vector_type(4)))  int      ivec4;

union FragH { v16h h; fvec4 f4[2]; ivec4 i4[2]; };

#define ATT_N 9216
#define ATT_C 64

// ---------------------------------------------------------------------------
// Generic 3x3 conv (zero pad 1) on fp32 VALU -- only for Ci=3 (input) and
// the final 16->1 conv. act: 0=none, 1=sigmoid. grid.y = co
// ---------------------------------------------------------------------------
__global__ void conv3x3_k(const float* __restrict__ in, const float* __restrict__ w,
                          const float* __restrict__ bias, float* __restrict__ out,
                          int Ci, int H, int W, int stride, int act) {
    __shared__ float wsh[96 * 9];
    const int co = blockIdx.y;
    const int nw = Ci * 9;
    for (int i = threadIdx.x; i < nw; i += blockDim.x)
        wsh[i] = w[(size_t)co * nw + i];
    __syncthreads();
    const int Ho = H / stride, Wo = W / stride;
    const int p = blockIdx.x * blockDim.x + threadIdx.x;
    if (p >= Ho * Wo) return;
    const int oy = p / Wo, ox = p % Wo;
    float acc = bias ? bias[co] : 0.0f;
    const int iy0 = oy * stride - 1, ix0 = ox * stride - 1;
    for (int ci = 0; ci < Ci; ++ci) {
        const float* ip = in + (size_t)ci * H * W;
        const float* wp = wsh + ci * 9;
#pragma unroll
        for (int ky = 0; ky < 3; ++ky) {
            int iy = iy0 + ky;
            if ((unsigned)iy >= (unsigned)H) continue;
#pragma unroll
            for (int kx = 0; kx < 3; ++kx) {
                int ix = ix0 + kx;
                if ((unsigned)ix < (unsigned)W)
                    acc = fmaf(ip[iy * W + ix], wp[ky * 3 + kx], acc);
            }
        }
    }
    if (act == 1) acc = 1.0f / (1.0f + __expf(-acc));
    out[((size_t)co * Ho + oy) * Wo + ox] = acc;
}

// ---------------------------------------------------------------------------
// Weight repack for WMMA conv: w[co][ci][3][3] f32 -> wpk[tap][co][ci] f16
// ---------------------------------------------------------------------------
__global__ void packw_k(const float* __restrict__ w, _Float16* __restrict__ wpk,
                        int Ci, int Co) {
    const int total = Co * Ci * 9;
    const int i = blockIdx.x * blockDim.x + threadIdx.x;
    if (i >= total) return;
    const int t  = i % 9;
    const int ci = (i / 9) % Ci;
    const int co = i / (9 * Ci);
    wpk[((size_t)t * Co + co) * Ci + ci] = (_Float16)w[i];
}

// ---------------------------------------------------------------------------
// Implicit-GEMM 3x3 conv with WMMA (templated on KC = Ci/32, TWO = Co==32).
//   in  : f16 HWC [Hin*Win][Ci],  wpk: f16 [9][Co][Ci],  out: f32 CHW [Co][HoWo]
// Weights staged in LDS once per WG; each wave owns TWO 16-pixel M-tiles so
// every B fragment (ds_load_b128) feeds 2 WMMAs; A fragments for the next
// (tap,chunk) are issued before the current chunk's WMMAs (software pipeline).
// Zero padding via clamped address + bitwise-AND mask (EXEC stays all-ones).
// ---------------------------------------------------------------------------
template <int KC, bool TWO>
__global__ __launch_bounds__(128) void conv_wmma_t(
        const _Float16* __restrict__ in, const _Float16* __restrict__ wpk,
        const float* __restrict__ bias, float* __restrict__ out,
        int Hin, int Win, int stride) {
    constexpr int Ci = KC * 32;
    constexpr int Co = TWO ? 32 : 16;
    __shared__ __align__(16) _Float16 wlds[9 * Co * Ci];

    // cooperative weight stage (bit copy through dwords)
    {
        const unsigned int* g = (const unsigned int*)wpk;
        unsigned int* l = (unsigned int*)wlds;
        constexpr int n32 = 9 * Co * Ci / 2;
        for (int i = threadIdx.x; i < n32; i += 128) l[i] = g[i];
    }
    __syncthreads();

    const int Ho = Hin / stride, Wo = Win / stride;
    const int HoWo = Ho * Wo;
    const int lane = threadIdx.x & 31;
    const int wv   = threadIdx.x >> 5;
    const int lr   = lane & 15;
    const int hi   = lane >> 4;
    const int p0   = (blockIdx.x * 4 + wv) * 32;
    const int pA = p0 + lr, pB = p0 + 16 + lr;
    const int oyA = pA / Wo, oxA = pA % Wo;
    const int oyB = pB / Wo, oxB = pB % Wo;

    v8f acc00, acc01, acc10, acc11;
    {
        const float b0 = bias[lr];
        const float b1 = TWO ? bias[16 + lr] : 0.0f;
#pragma unroll
        for (int v = 0; v < 8; ++v) { acc00[v] = b0; acc10[v] = b0; acc01[v] = b1; acc11[v] = b1; }
    }

    auto loadA = [&](int oy, int ox, int t, int cc, FragH& a) {
        const int dy = t / 3, dx = t % 3;
        const int iy = oy * stride - 1 + dy;
        const int ix = ox * stride - 1 + dx;
        const bool inr = ((unsigned)iy < (unsigned)Hin) && ((unsigned)ix < (unsigned)Win);
        const int iyc = iy < 0 ? 0 : (iy >= Hin ? Hin - 1 : iy);
        const int ixc = ix < 0 ? 0 : (ix >= Win ? Win - 1 : ix);
        const fvec4* ar = (const fvec4*)(in + ((size_t)iyc * Win + ixc) * Ci + (cc << 5));
        a.f4[0] = ar[2 * hi];
        a.f4[1] = ar[2 * hi + 1];
        const int msk = inr ? -1 : 0;          // bit-exact zero, no NaN hazard
        a.i4[0] &= msk;
        a.i4[1] &= msk;
    };

    FragH a0c, a1c, a0n, a1n;
    loadA(oyA, oxA, 0, 0, a0c);
    loadA(oyB, oxB, 0, 0, a1c);

    constexpr int NKCH = 9 * KC;
#pragma unroll
    for (int k = 0; k < NKCH; ++k) {
        const int nk = (k + 1 < NKCH) ? k + 1 : k;
        loadA(oyA, oxA, nk / KC, nk % KC, a0n);      // prefetch next chunk
        loadA(oyB, oxB, nk / KC, nk % KC, a1n);
        const int t = k / KC, cc = k % KC;
        const _Float16* wb = wlds + (t * Co) * Ci + (cc << 5);
        FragH b0;
        {
            const fvec4* br = (const fvec4*)(wb + lr * Ci);
            b0.f4[0] = br[2 * hi]; b0.f4[1] = br[2 * hi + 1];
        }
        acc00 = __builtin_amdgcn_wmma_f32_16x16x32_f16(false, a0c.h, false, b0.h, (short)0, acc00, false, false);
        acc10 = __builtin_amdgcn_wmma_f32_16x16x32_f16(false, a1c.h, false, b0.h, (short)0, acc10, false, false);
        if (TWO) {
            FragH b1;
            const fvec4* br = (const fvec4*)(wb + (16 + lr) * Ci);
            b1.f4[0] = br[2 * hi]; b1.f4[1] = br[2 * hi + 1];
            acc01 = __builtin_amdgcn_wmma_f32_16x16x32_f16(false, a0c.h, false, b1.h, (short)0, acc01, false, false);
            acc11 = __builtin_amdgcn_wmma_f32_16x16x32_f16(false, a1c.h, false, b1.h, (short)0, acc11, false, false);
        }
        a0c = a0n; a1c = a1n;
    }

#pragma unroll
    for (int v = 0; v < 8; ++v) {
        const int r = v + hi * 8;
        out[(size_t)lr * HoWo + p0 + r]      = acc00[v];
        out[(size_t)lr * HoWo + p0 + 16 + r] = acc10[v];
    }
    if (TWO) {
#pragma unroll
        for (int v = 0; v < 8; ++v) {
            const int r = v + hi * 8;
            out[(size_t)(16 + lr) * HoWo + p0 + r]      = acc01[v];
            out[(size_t)(16 + lr) * HoWo + p0 + 16 + r] = acc11[v];
        }
    }
}

// ---------------------------------------------------------------------------
// 1x1 conv (QKV projections). grid.y = co
// ---------------------------------------------------------------------------
__global__ void conv1x1_k(const float* __restrict__ in, const float* __restrict__ w,
                          const float* __restrict__ bias, float* __restrict__ out,
                          int Ci, int HWsz) {
    __shared__ float wsh[64];
    const int co = blockIdx.y;
    if ((int)threadIdx.x < Ci) wsh[threadIdx.x] = w[(size_t)co * Ci + threadIdx.x];
    __syncthreads();
    const int p = blockIdx.x * blockDim.x + threadIdx.x;
    if (p >= HWsz) return;
    float acc = bias[co];
    for (int ci = 0; ci < Ci; ++ci)
        acc = fmaf(wsh[ci], in[(size_t)ci * HWsz + p], acc);
    out[(size_t)co * HWsz + p] = acc;
}

// ---------------------------------------------------------------------------
// Instance-norm statistics: one workgroup per channel
// ---------------------------------------------------------------------------
__global__ void inorm_stats_k(const float* __restrict__ x, float* __restrict__ mean,
                              float* __restrict__ rstd, int HWsz) {
    __shared__ float ssum[256];
    __shared__ float ssq[256];
    const int c = blockIdx.x;
    const int tid = threadIdx.x;
    float s = 0.0f, q = 0.0f;
    for (int i = tid; i < HWsz; i += 256) {
        float v = x[(size_t)c * HWsz + i];
        s += v; q += v * v;
    }
    ssum[tid] = s; ssq[tid] = q;
    __syncthreads();
    for (int off = 128; off > 0; off >>= 1) {
        if (tid < off) { ssum[tid] += ssum[tid + off]; ssq[tid] += ssq[tid + off]; }
        __syncthreads();
    }
    if (tid == 0) {
        float m = ssum[0] / (float)HWsz;
        float v = ssq[0] / (float)HWsz - m * m;
        mean[c] = m;
        rstd[c] = rsqrtf(v + 1e-5f);
    }
}

// ---------------------------------------------------------------------------
// Instance-norm apply (+optional residual) + leaky ReLU(0.2), in-place safe.
// Optionally mirrors the result to an f16 HWC buffer for the next WMMA conv.
// ---------------------------------------------------------------------------
__global__ void inorm_apply_k(const float* __restrict__ x, const float* __restrict__ mean,
                              const float* __restrict__ rstd, const float* __restrict__ res,
                              float* __restrict__ out, int HWsz, int total,
                              _Float16* __restrict__ ohwc, int Cn) {
    const int i = blockIdx.x * blockDim.x + threadIdx.x;
    if (i >= total) return;
    const int c = i / HWsz;
    const int p = i - c * HWsz;
    float v = (x[i] - mean[c]) * rstd[c];
    if (res) v += res[i];
    v = v > 0.0f ? v : 0.2f * v;
    out[i] = v;
    if (ohwc) ohwc[(size_t)p * Cn + c] = (_Float16)v;
}

// ---------------------------------------------------------------------------
// Pack normalized+lrelu fp32 [C][N] into f16 [N][C] (Q/K; Q gets temp/sqrt(C))
// ---------------------------------------------------------------------------
__global__ void pack_nc_k(const float* __restrict__ x, const float* __restrict__ mean,
                          const float* __restrict__ rstd, _Float16* __restrict__ out,
                          const float* __restrict__ tscale, float smul, int Cn, int Nn) {
    const int i = blockIdx.x * blockDim.x + threadIdx.x;
    if (i >= Cn * Nn) return;
    const int c = i / Nn, n = i % Nn;
    float v = (x[i] - mean[c]) * rstd[c];
    v = v > 0.0f ? v : 0.2f * v;
    const float s = smul * (tscale ? tscale[0] : 1.0f);
    out[(size_t)n * Cn + c] = (_Float16)(v * s);
}

// Pack to f16 keeping [C][N] layout (V)
__global__ void pack_cn_k(const float* __restrict__ x, const float* __restrict__ mean,
                          const float* __restrict__ rstd, _Float16* __restrict__ out,
                          int Cn, int Nn) {
    const int i = blockIdx.x * blockDim.x + threadIdx.x;
    if (i >= Cn * Nn) return;
    const int c = i / Nn;
    float v = (x[i] - mean[c]) * rstd[c];
    v = v > 0.0f ? v : 0.2f * v;
    out[i] = (_Float16)v;
}

// ---------------------------------------------------------------------------
// Flash attention with WMMA f32_16x16x32_f16.
//   Qh: [N,64] f16 (pre-scaled), Kh: [N,64] f16, Vh: [64,N] f16, agg: [64,N] f32
// ---------------------------------------------------------------------------
__global__ __launch_bounds__(128) void flash_attn_k(const _Float16* __restrict__ Qh,
                                                    const _Float16* __restrict__ Kh,
                                                    const _Float16* __restrict__ Vh,
                                                    float* __restrict__ agg) {
    __shared__ __align__(16) float    sS[4][16][32];
    __shared__ __align__(16) _Float16 sP[4][16][32];
    __shared__ float sAlpha[4][16];
    __shared__ float sL[4][16];

    const int lane = threadIdx.x & 31;
    const int wv   = threadIdx.x >> 5;
    const int lr   = lane & 15;
    const int hi   = lane >> 4;
    const int q0   = blockIdx.x * 64 + wv * 16;

    FragH aq0, aq1;
    {
        const fvec4* qr = (const fvec4*)(Qh + (size_t)(q0 + lr) * 64);
        aq0.f4[0] = qr[2 * hi];     aq0.f4[1] = qr[2 * hi + 1];
        aq1.f4[0] = qr[4 + 2 * hi]; aq1.f4[1] = qr[4 + 2 * hi + 1];
    }

    v8f o0, o1, o2, o3;
#pragma unroll
    for (int i = 0; i < 8; ++i) { o0[i] = 0.f; o1[i] = 0.f; o2[i] = 0.f; o3[i] = 0.f; }
    float m_run = -3.0e38f, l_run = 0.0f;   // valid in lanes 0..15

    for (int blk = 0; blk < ATT_N; blk += 32) {
        // ---- issue ALL global fragment loads for this block up front ----
        FragH bk00, bk01, bk10, bk11;
        {
            const fvec4* k0 = (const fvec4*)(Kh + (size_t)(blk + lr) * 64);
            bk00.f4[0] = k0[2 * hi];     bk00.f4[1] = k0[2 * hi + 1];
            bk01.f4[0] = k0[4 + 2 * hi]; bk01.f4[1] = k0[4 + 2 * hi + 1];
            const fvec4* k1 = (const fvec4*)(Kh + (size_t)(blk + 16 + lr) * 64);
            bk10.f4[0] = k1[2 * hi];     bk10.f4[1] = k1[2 * hi + 1];
            bk11.f4[0] = k1[4 + 2 * hi]; bk11.f4[1] = k1[4 + 2 * hi + 1];
        }
        FragH bv0, bv1, bv2, bv3;
        {
            const fvec4* v0 = (const fvec4*)(Vh + (size_t)(0 + lr) * ATT_N + blk);
            bv0.f4[0] = v0[2 * hi]; bv0.f4[1] = v0[2 * hi + 1];
            const fvec4* v1 = (const fvec4*)(Vh + (size_t)(16 + lr) * ATT_N + blk);
            bv1.f4[0] = v1[2 * hi]; bv1.f4[1] = v1[2 * hi + 1];
            const fvec4* v2 = (const fvec4*)(Vh + (size_t)(32 + lr) * ATT_N + blk);
            bv2.f4[0] = v2[2 * hi]; bv2.f4[1] = v2[2 * hi + 1];
            const fvec4* v3 = (const fvec4*)(Vh + (size_t)(48 + lr) * ATT_N + blk);
            bv3.f4[0] = v3[2 * hi]; bv3.f4[1] = v3[2 * hi + 1];
        }
        if (blk + 32 < ATT_N) {
            __builtin_prefetch(Kh + (size_t)(blk + 32 + lr) * 64, 0, 1);
            __builtin_prefetch(Kh + (size_t)(blk + 48 + lr) * 64, 0, 1);
            __builtin_prefetch(Vh + (size_t)(0  + lr) * ATT_N + blk + 32, 0, 1);
            __builtin_prefetch(Vh + (size_t)(32 + lr) * ATT_N + blk + 32, 0, 1);
        }

        v8f s0, s1;
#pragma unroll
        for (int i = 0; i < 8; ++i) { s0[i] = 0.f; s1[i] = 0.f; }
        s0 = __builtin_amdgcn_wmma_f32_16x16x32_f16(false, aq0.h, false, bk00.h, (short)0, s0, false, false);
        s0 = __builtin_amdgcn_wmma_f32_16x16x32_f16(false, aq1.h, false, bk01.h, (short)0, s0, false, false);
        s1 = __builtin_amdgcn_wmma_f32_16x16x32_f16(false, aq0.h, false, bk10.h, (short)0, s1, false, false);
        s1 = __builtin_amdgcn_wmma_f32_16x16x32_f16(false, aq1.h, false, bk11.h, (short)0, s1, false, false);

#pragma unroll
        for (int v = 0; v < 8; ++v) {
            sS[wv][v + hi * 8][lr]      = s0[v];
            sS[wv][v + hi * 8][16 + lr] = s1[v];
        }
        __syncthreads();

        if (lane < 16) {
            float tm = -3.0e38f;
            for (int j = 0; j < 32; ++j) tm = fmaxf(tm, sS[wv][lr][j]);
            float mn    = fmaxf(m_run, tm);
            float alpha = __expf(m_run - mn);
            float ssum  = 0.0f;
            for (int j = 0; j < 32; ++j) {
                float pv = __expf(sS[wv][lr][j] - mn);
                ssum += pv;
                sP[wv][lr][j] = (_Float16)pv;
            }
            l_run = l_run * alpha + ssum;
            m_run = mn;
            sAlpha[wv][lr] = alpha;
        }
        __syncthreads();

        float av[8];
#pragma unroll
        for (int v = 0; v < 8; ++v) av[v] = sAlpha[wv][v + hi * 8];
#pragma unroll
        for (int v = 0; v < 8; ++v) { o0[v] *= av[v]; o1[v] *= av[v]; o2[v] *= av[v]; o3[v] *= av[v]; }

        FragH pf;
        {
            const fvec4* pr = (const fvec4*)&sP[wv][lr][0];
            pf.f4[0] = pr[2 * hi];
            pf.f4[1] = pr[2 * hi + 1];
        }
        o0 = __builtin_amdgcn_wmma_f32_16x16x32_f16(false, pf.h, false, bv0.h, (short)0, o0, false, false);
        o1 = __builtin_amdgcn_wmma_f32_16x16x32_f16(false, pf.h, false, bv1.h, (short)0, o1, false, false);
        o2 = __builtin_amdgcn_wmma_f32_16x16x32_f16(false, pf.h, false, bv2.h, (short)0, o2, false, false);
        o3 = __builtin_amdgcn_wmma_f32_16x16x32_f16(false, pf.h, false, bv3.h, (short)0, o3, false, false);
        __syncthreads();
    }

    if (lane < 16) sL[wv][lr] = l_run;
    __syncthreads();
#pragma unroll
    for (int v = 0; v < 8; ++v) {
        const float inv = 1.0f / sL[wv][v + hi * 8];
        const int qq = q0 + v + hi * 8;
        agg[(size_t)(0  + lr) * ATT_N + qq] = o0[v] * inv;
        agg[(size_t)(16 + lr) * ATT_N + qq] = o1[v] * inv;
        agg[(size_t)(32 + lr) * ATT_N + qq] = o2[v] * inv;
        agg[(size_t)(48 + lr) * ATT_N + qq] = o3[v] * inv;
    }
}

// ---------------------------------------------------------------------------
// Per-channel avg + max pooling (one WG per channel)
// ---------------------------------------------------------------------------
__global__ void pool_k(const float* __restrict__ x, float* __restrict__ avg,
                       float* __restrict__ mx, int HWsz) {
    __shared__ float ss[256];
    __shared__ float sm[256];
    const int c = blockIdx.x;
    const int tid = threadIdx.x;
    float s = 0.0f, m = -3.0e38f;
    for (int i = tid; i < HWsz; i += 256) {
        float v = x[(size_t)c * HWsz + i];
        s += v; m = fmaxf(m, v);
    }
    ss[tid] = s; sm[tid] = m;
    __syncthreads();
    for (int off = 128; off > 0; off >>= 1) {
        if (tid < off) { ss[tid] += ss[tid + off]; sm[tid] = fmaxf(sm[tid], sm[tid + off]); }
        __syncthreads();
    }
    if (tid == 0) { avg[c] = ss[0] / (float)HWsz; mx[c] = sm[0]; }
}

// ---------------------------------------------------------------------------
// Channel-attention MLP: sigmoid(fc2(relu(fc1 avg)) + fc2(relu(fc1 mx)))
// ---------------------------------------------------------------------------
__global__ void ca_scale_k(const float* __restrict__ avg, const float* __restrict__ mx,
                           const float* __restrict__ fc1w, const float* __restrict__ fc2w,
                           float* __restrict__ cs) {
    __shared__ float h[8];
    const int tid = threadIdx.x;
    if (tid < 8) {
        float a = 0.0f, m = 0.0f;
        for (int j = 0; j < 64; ++j) {
            a = fmaf(fc1w[tid * 64 + j], avg[j], a);
            m = fmaf(fc1w[tid * 64 + j], mx[j],  m);
        }
        h[tid] = fmaxf(a, 0.0f) + fmaxf(m, 0.0f);
    }
    __syncthreads();
    float s = 0.0f;
    for (int j = 0; j < 8; ++j) s = fmaf(fc2w[tid * 8 + j], h[j], s);
    cs[tid] = 1.0f / (1.0f + __expf(-s));
}

// ---------------------------------------------------------------------------
// 2x bilinear upsample (align_corners=True) with optional per-channel scale.
// Writes f32 CHW (optional) and/or f16 HWC (optional, with channel offset).
// ---------------------------------------------------------------------------
__global__ void upsample2x_k(const float* __restrict__ in, float* __restrict__ out,
                             int Cn, int Hin, int Win, const float* __restrict__ cscale,
                             _Float16* __restrict__ ohwc, int Cs, int coff) {
    const int Hout = 2 * Hin, Wout = 2 * Win;
    const int i = blockIdx.x * blockDim.x + threadIdx.x;
    if (i >= Cn * Hout * Wout) return;
    const int x = i % Wout;
    const int y = (i / Wout) % Hout;
    const int c = i / (Wout * Hout);
    float ry = (float)y * ((float)(Hin - 1) / (float)(Hout - 1));
    int y0 = (int)ry; float ty = ry - (float)y0;
    int y1 = (y0 + 1 < Hin) ? y0 + 1 : Hin - 1;
    float rx = (float)x * ((float)(Win - 1) / (float)(Wout - 1));
    int x0 = (int)rx; float tx = rx - (float)x0;
    int x1 = (x0 + 1 < Win) ? x0 + 1 : Win - 1;
    const float* ip = in + (size_t)c * Hin * Win;
    float a = ip[y0 * Win + x0] * (1.0f - ty) + ip[y1 * Win + x0] * ty;
    float b = ip[y0 * Win + x1] * (1.0f - ty) + ip[y1 * Win + x1] * ty;
    float v = a * (1.0f - tx) + b * tx;
    if (cscale) v *= cscale[c];
    if (out) out[i] = v;
    if (ohwc) ohwc[(size_t)(y * Wout + x) * Cs + coff + c] = (_Float16)v;
}

// out = f*m + base (f32 CHW), mirrored to f16 HWC
__global__ void dyn_k(const float* __restrict__ f, const float* __restrict__ m,
                      const float* __restrict__ base, float* __restrict__ out, int n,
                      int HWsz, _Float16* __restrict__ ohwc, int Cn) {
    const int i = blockIdx.x * blockDim.x + threadIdx.x;
    if (i >= n) return;
    const float v = fmaf(f[i], m[i], base[i]);
    out[i] = v;
    const int c = i / HWsz;
    const int p = i - c * HWsz;
    ohwc[(size_t)p * Cn + c] = (_Float16)v;
}

// ---------------------------------------------------------------------------
// Host orchestration
// ---------------------------------------------------------------------------
extern "C" void kernel_launch(void* const* d_in, const int* in_sizes, int n_in,
                              void* d_out, int out_size, void* d_ws, size_t ws_size,
                              hipStream_t stream) {
    (void)in_sizes; (void)n_in; (void)out_size; (void)ws_size;
    auto P = [&](int i) -> const float* { return (const float*)d_in[i]; };

    const int H = 192, Wd = 192, HW = H * Wd;   // 36864
    const int h2 = 96, hw = h2 * h2;            // 9216

    float* base  = (float*)d_ws;
    float* A     = base;                          // 32*HW f32
    float* Bb    = A     + (size_t)32 * HW;       // 32*HW f32 (also agg 64*hw)
    float* Cc    = Bb    + (size_t)32 * HW;       // 32*HW f32
    float* FCF   = Cc    + (size_t)32 * HW;       // feat_c_full f32
    float* FEATC = FCF   + (size_t)32 * HW;       // 32*hw f32
    float* FEATS = FEATC + (size_t)32 * hw;
    float* MEAN  = FEATS + (size_t)32 * hw;       // 128
    float* RSTD  = MEAN + 128;
    float* AVG   = RSTD + 128;
    float* MX    = AVG + 64;
    float* CS    = MX + 64;
    _Float16* QH     = (_Float16*)(CS + 64);      // [N,64]
    _Float16* KH     = QH + (size_t)64 * hw;      // [N,64]
    _Float16* VH     = KH + (size_t)64 * hw;      // [64,N]
    _Float16* H1     = VH + (size_t)64 * hw;      // [HW][32] f16 HWC
    _Float16* H2     = H1 + (size_t)32 * HW;      // [HW][32]
    _Float16* COMB16 = H2 + (size_t)32 * HW;      // [HW][96]
    _Float16* FCF16  = COMB16 + (size_t)96 * HW;  // [HW][32]
    _Float16* WPK    = FCF16 + (size_t)32 * HW;   // packed weights (<=27648)

    auto conv3 = [&](const float* in, int wi, int bi, float* out,
                     int Ci, int Co, int Hh, int Ww, int stride, int act) {
        int Ho = Hh / stride, Wo = Ww / stride;
        dim3 g((Ho * Wo + 255) / 256, Co);
        conv3x3_k<<<g, dim3(256), 0, stream>>>(in, P(wi), P(bi), out, Ci, Hh, Ww, stride, act);
    };
    auto wconv = [&](const _Float16* inh, int wi, int bi, float* out,
                     int Ci, int Co, int Hin, int Win, int stride) {
        int tot = Co * Ci * 9;
        packw_k<<<dim3((tot + 255) / 256), dim3(256), 0, stream>>>(P(wi), WPK, Ci, Co);
        int Ho = Hin / stride, Wo = Win / stride;
        dim3 g(Ho * Wo / 128);
        if (Ci == 96)
            conv_wmma_t<3, true ><<<g, dim3(128), 0, stream>>>(inh, WPK, P(bi), out, Hin, Win, stride);
        else if (Co == 32)
            conv_wmma_t<1, true ><<<g, dim3(128), 0, stream>>>(inh, WPK, P(bi), out, Hin, Win, stride);
        else
            conv_wmma_t<1, false><<<g, dim3(128), 0, stream>>>(inh, WPK, P(bi), out, Hin, Win, stride);
    };
    auto instnorm = [&](float* x, int Cn, int HWsz, const float* res, _Float16* ohwc) {
        inorm_stats_k<<<dim3(Cn), dim3(256), 0, stream>>>(x, MEAN, RSTD, HWsz);
        int total = Cn * HWsz;
        inorm_apply_k<<<dim3((total + 255) / 256), dim3(256), 0, stream>>>(x, MEAN, RSTD, res, x, HWsz, total, ohwc, Cn);
    };
    auto encode = [&](const float* img, int pb, float* feat) {
        conv3(img, pb + 0, pb + 1, A, 3, 32, H, Wd, 1, 0);
        instnorm(A, 32, HW, nullptr, H1);                                 // h1
        wconv(H1, pb + 2, pb + 3, Bb, 32, 32, H, Wd, 1);
        instnorm(Bb, 32, HW, nullptr, H2);                                // r1
        wconv(H2, pb + 4, pb + 5, Cc, 32, 32, H, Wd, 1);
        instnorm(Cc, 32, HW, A, H1);                                      // r2 (+h1 res)
        wconv(H1, pb + 6, pb + 7, feat, 32, 32, H, Wd, 2);                // stride-2
        instnorm(feat, 32, hw, nullptr, nullptr);
    };

    // Encoders
    encode(P(0), 2, FEATC);
    encode(P(1), 10, FEATS);

    // Q/K/V projections -> f16 fragments
    auto proj = [&](const float* feat, int wi, int bi) {
        dim3 g((hw + 255) / 256, 64);
        conv1x1_k<<<g, dim3(256), 0, stream>>>(feat, P(wi), P(bi), A, 32, hw);
        inorm_stats_k<<<dim3(64), dim3(256), 0, stream>>>(A, MEAN, RSTD, hw);
    };
    const int tot64 = 64 * hw;
    proj(FEATC, 18, 19);
    pack_nc_k<<<dim3((tot64 + 255) / 256), dim3(256), 0, stream>>>(A, MEAN, RSTD, QH, P(24), 0.125f, 64, hw);
    proj(FEATS, 20, 21);
    pack_nc_k<<<dim3((tot64 + 255) / 256), dim3(256), 0, stream>>>(A, MEAN, RSTD, KH, nullptr, 1.0f, 64, hw);
    proj(FEATS, 22, 23);
    pack_cn_k<<<dim3((tot64 + 255) / 256), dim3(256), 0, stream>>>(A, MEAN, RSTD, VH, 64, hw);

    // Flash attention (WMMA): agg[64, hw] -> Bb
    flash_attn_k<<<dim3(hw / 64), dim3(128), 0, stream>>>(QH, KH, VH, Bb);

    // Channel attention
    pool_k<<<dim3(64), dim3(256), 0, stream>>>(Bb, AVG, MX, hw);
    ca_scale_k<<<dim3(1), dim3(64), 0, stream>>>(AVG, MX, P(25), P(26), CS);

    // Upsample: feat_c -> FCF (f32 + f16 HWC) and into COMB16[.,0:32];
    // agg (channel-scaled) -> COMB16[.,32:96]
    upsample2x_k<<<dim3((32 * HW + 255) / 256), dim3(256), 0, stream>>>(FEATC, FCF, 32, h2, h2, nullptr, FCF16, 32, 0);
    upsample2x_k<<<dim3((32 * HW + 255) / 256), dim3(256), 0, stream>>>(FEATC, nullptr, 32, h2, h2, nullptr, COMB16, 96, 0);
    upsample2x_k<<<dim3((64 * HW + 255) / 256), dim3(256), 0, stream>>>(Bb, nullptr, 64, h2, h2, CS, COMB16, 96, 32);

    // Merge decoder (all WMMA convs)
    wconv(COMB16, 27, 28, A, 96, 32, H, Wd, 1);  instnorm(A, 32, HW, nullptr, H1);   // m1
    wconv(H1,     29, 30, Bb, 32, 32, H, Wd, 1); instnorm(Bb, 32, HW, nullptr, H2);  // mr1
    wconv(H2,     31, 32, Cc, 32, 32, H, Wd, 1); instnorm(Cc, 32, HW, A, H1);        // mr2 (+m1)
    wconv(H1,     33, 34, Bb, 32, 32, H, Wd, 1); instnorm(Bb, 32, HW, nullptr, nullptr); // m
    wconv(FCF16,  35, 36, A, 32, 32, H, Wd, 1);  instnorm(A, 32, HW, nullptr, nullptr);  // f
    dyn_k<<<dim3((32 * HW + 255) / 256), dim3(256), 0, stream>>>(A, Bb, FCF, Cc, 32 * HW, HW, H1, 32);
    wconv(H1, 37, 38, A, 32, 16, H, Wd, 1);      instnorm(A, 16, HW, nullptr, nullptr);  // o1
    conv3(A, 39, 40, (float*)d_out, 16, 1, H, Wd, 1, 1);                                 // sigmoid
}
